// WorkingMemory_18640158065229
// MI455X (gfx1250) — matrix operand order repaired and checked by reading, not compiled, not run on hardware
//
#include <hip/hip_runtime.h>
#include <hip/hip_bf16.h>
#include <math.h>

// ---------------- types ----------------
typedef __attribute__((ext_vector_type(4)))  __bf16 v4bf;
typedef __attribute__((ext_vector_type(8)))  __bf16 v8bf;
typedef __attribute__((ext_vector_type(16))) __bf16 v16bf;
typedef __attribute__((ext_vector_type(8)))  float  v8f;
typedef __attribute__((ext_vector_type(2)))  float  v2f;

#define Dm 1024
#define CAP 16
#define Bn 32768
#define MT 4                                  // 16-row tiles per block (M = 64)
#define NWAVE 4                               // waves per block, each owns 1/4 of N-chunks
#define ROWS_PER_BLOCK (MT * 16)              // 64
#define NBLOCKS (Bn / ROWS_PER_BLOCK)         // 512
// LDS layout (bytes)
#define OFF_XT 0                              // bf16 [64][1024] = 131072
#define OFF_RETR 131072                       // bf16 [64][1024] = 131072
#define OFF_ATTN 262144                       // f32  [64][16]   = 4096
#define OFF_RSQ 266240                        // f32  [4][64]    = 1024
#define OFF_RHS 267264                        // f32  [4][64]    = 1024
#define OFF_RHQ 268288                        // f32  [4][64]    = 1024
#define LDS_BYTES 269312

// ---------------- fragment helpers ----------------
__device__ __forceinline__ v16bf combine16(v8bf lo, v8bf hi) {
  v16bf r;
#pragma unroll
  for (int i = 0; i < 8; ++i) { r[i] = lo[i]; r[i + 8] = hi[i]; }
  return r;
}
// A (16x32 bf16): lane m=ln; lanes<16 hold K = kc+{0..7,16..23}, lanes>=16 K = kc+{8..15,24..31}
__device__ __forceinline__ v16bf load_afrag(const __bf16* rowp, int kbase) {
  v8bf lo = *(const v8bf*)(rowp + kbase);
  v8bf hi = *(const v8bf*)(rowp + kbase + 16);
  return combine16(lo, hi);
}
// B (32x16 bf16): lane n=ln; lanes<16 hold K = kc+{0..7,16..23}, lanes>=16 K = kc+{8..15,24..31}
__device__ __forceinline__ v16bf load_bfrag(const __bf16* p) {
  v8bf lo = *(const v8bf*)(p);
  v8bf hi = *(const v8bf*)(p + 16);
  return combine16(lo, hi);
}
__device__ __forceinline__ float grp16_sum(float v) {
  v += __shfl_xor(v, 1); v += __shfl_xor(v, 2);
  v += __shfl_xor(v, 4); v += __shfl_xor(v, 8);
  return v;
}
__device__ __forceinline__ float grp16_max(float v) {
  v = fmaxf(v, __shfl_xor(v, 1)); v = fmaxf(v, __shfl_xor(v, 2));
  v = fmaxf(v, __shfl_xor(v, 4)); v = fmaxf(v, __shfl_xor(v, 8));
  return v;
}

// ---------------- prep kernels ----------------
__global__ void wm_prep1(const float* __restrict__ buf, const float* __restrict__ Wk,
                         const float* __restrict__ Wv, float* __restrict__ k_ws,
                         float* __restrict__ v_ws) {
  int gid = blockIdx.x * 256 + threadIdx.x;          // 0..32767
  int mat = gid >> 14;
  int j = (gid >> 10) & 15;
  int d = gid & 1023;
  const float* W = mat ? Wv : Wk;
  float* o = mat ? v_ws : k_ws;
  const float* br = buf + j * Dm;
  const float* wr = W + d * Dm;
  float s = 0.f;
#pragma unroll 4
  for (int i = 0; i < Dm; ++i) s = fmaf(br[i], wr[i], s);
  o[j * Dm + d] = s;
}

__global__ void wm_prep2(const float* __restrict__ buf_age, const float* __restrict__ buf_freq,
                         const float* __restrict__ buf_imp, float* __restrict__ k_ws,
                         float* __restrict__ salc) {
  __shared__ float norms[16];
  __shared__ float fm;
  int t = threadIdx.x;
  if (t == 0) {
    float m = buf_freq[0];
    for (int j = 1; j < CAP; ++j) m = fmaxf(m, buf_freq[j]);
    fm = m;
  }
  if (t < 16) {
    float s = 0.f;
    const float* kr = k_ws + t * Dm;
    for (int i = 0; i < Dm; ++i) s = fmaf(kr[i], kr[i], s);
    norms[t] = fmaxf(sqrtf(s), 1e-12f);
  }
  __syncthreads();
  if (t < 16) {
    float rec = __expf(-buf_age[t] / 200.0f);
    float fn = logf(buf_freq[t] + 1.0f) / (logf(fm + 2.0f) + 1e-8f);
    salc[t] = 0.20f * rec + 0.15f * fn + 0.10f * buf_imp[t] + 0.10f * 1.0f;
  }
  __syncthreads();
  for (int i = t; i < CAP * Dm; i += 256) k_ws[i] = k_ws[i] / norms[i >> 10];
}

// M = k_norm @ Wq  -> bf16 [16][1024]
__global__ void wm_prep3(const float* __restrict__ k_ws, const float* __restrict__ Wq,
                         __bf16* __restrict__ Mb) {
  int gid = blockIdx.x * 256 + threadIdx.x;          // 0..16383
  int j = gid >> 10;
  int kc = gid & 1023;
  float s = 0.f;
#pragma unroll 4
  for (int d = 0; d < Dm; ++d) s = fmaf(k_ws[j * Dm + d], Wq[d * Dm + kc], s);
  Mb[j * Dm + kc] = (__bf16)s;
}

// weights fp32 -> bf16
__global__ void wm_conv(const float* __restrict__ Wq, const float* __restrict__ Wg,
                        const float* __restrict__ Wo, __bf16* __restrict__ Wqb,
                        __bf16* __restrict__ Wgb, __bf16* __restrict__ Wob) {
  long i0 = (long)blockIdx.x * 256 + threadIdx.x;
  long stride = (long)gridDim.x * 256;
  for (long j = i0; j < (long)Dm * Dm; j += stride) Wqb[j] = (__bf16)Wq[j];
  for (long j = i0; j < (long)Dm * 2 * Dm; j += stride) {
    Wgb[j] = (__bf16)Wg[j];
    Wob[j] = (__bf16)Wo[j];
  }
}

// ---------------- main fused kernel: 64 rows per block, 4 cooperative waves ----------------
__global__ void __launch_bounds__(NWAVE * 32)
wm_main(const float* __restrict__ x, const float* __restrict__ v_ws,
        const float* __restrict__ salc, const __bf16* __restrict__ Mb,
        const __bf16* __restrict__ Wqb, const __bf16* __restrict__ Wgb,
        const __bf16* __restrict__ Wob, const float* __restrict__ bg,
        const float* __restrict__ bo, float* __restrict__ out,
        float* __restrict__ stats) {
  extern __shared__ __align__(16) char smem[];
  __bf16* xt    = (__bf16*)(smem + OFF_XT);      // [64][1024] bf16
  __bf16* retr  = (__bf16*)(smem + OFF_RETR);    // [64][1024] bf16
  float*  attnL = (float*)(smem + OFF_ATTN);     // [64][16]
  float*  rsq   = (float*)(smem + OFF_RSQ);      // [4][64]
  float*  rhs   = (float*)(smem + OFF_RHS);      // [4][64]
  float*  rhq   = (float*)(smem + OFF_RHQ);      // [4][64]

  const int tid = threadIdx.x;
  const int lane = tid & 31;
  const int wid = tid >> 5;                      // wave 0..3, owns n0 = wid*16 step 64
  const int ln = lane & 15;
  const int lh = lane >> 4;
  const long R0 = (long)blockIdx.x * ROWS_PER_BLOCK;
  const int akoff = 8 * lh;
  const int bkoff = 8 * lh;

  // ---- phase 0: cooperative x tile -> LDS bf16 ----
#pragma unroll 4
  for (int it = 0; it < 128; ++it) {
    int i = tid + it * 128;                      // 16384 float4s
    int row = i >> 8;
    int c4 = (i & 255) << 2;
    float4 xv = *(const float4*)(x + (R0 + row) * Dm + c4);
    v4bf b;
    b[0] = (__bf16)xv.x; b[1] = (__bf16)xv.y; b[2] = (__bf16)xv.z; b[3] = (__bf16)xv.w;
    *(v4bf*)(xt + row * Dm + c4) = b;
  }
  __syncthreads();

  // ---- phase 1: sim logits t = x @ M.T ; wave w owns tile w ----
  const __bf16* myxrow = xt + (wid * 16 + ln) * Dm;
  v8f tacc = {};
#pragma unroll 4
  for (int kc = 0; kc < Dm; kc += 32) {
    v16bf a = load_afrag(myxrow, kc + akoff);
    v16bf b = load_bfrag(Mb + ln * Dm + kc + bkoff);
    tacc = __builtin_amdgcn_wmma_f32_16x16x32_bf16(false, a, false, b, (short)0, tacc, false, false);
  }

  // ---- phase 2: q sum-of-squares partials (each wave does 1/4 of N, all 4 tiles) ----
  {
    v8f sq[MT];
#pragma unroll
    for (int tt = 0; tt < MT; ++tt) sq[tt] = (v8f){};
    for (int n0 = wid * 16; n0 < Dm; n0 += 64) {
      v8f acc[MT];
#pragma unroll
      for (int tt = 0; tt < MT; ++tt) acc[tt] = (v8f){};
      const __bf16* wq = Wqb + (n0 + ln) * Dm;
#pragma unroll 2
      for (int kc = 0; kc < Dm; kc += 32) {
        v16bf b = load_bfrag(wq + kc + bkoff);
#pragma unroll
        for (int tt = 0; tt < MT; ++tt) {
          v16bf a = load_afrag(xt + (tt * 16 + ln) * Dm, kc + akoff);
          acc[tt] = __builtin_amdgcn_wmma_f32_16x16x32_bf16(false, a, false, b, (short)0, acc[tt], false, false);
        }
      }
#pragma unroll
      for (int tt = 0; tt < MT; ++tt) sq[tt] += acc[tt] * acc[tt];
    }
#pragma unroll
    for (int tt = 0; tt < MT; ++tt)
#pragma unroll
      for (int r = 0; r < 8; ++r) {
        float s = grp16_sum(sq[tt][r]);
        if (ln == 0) rsq[wid * 64 + tt * 16 + r + 8 * lh] = s;
      }
  }
  __syncthreads();

  // ---- phase 3: softmax over 16 slots; wave w owns tile w ----
  {
    float cj = salc[ln];
#pragma unroll
    for (int r = 0; r < 8; ++r) {
      int row = wid * 16 + r + 8 * lh;
      float s = rsq[row] + rsq[64 + row] + rsq[128 + row] + rsq[192 + row];
      float nrm = fmaxf(sqrtf(s), 1e-12f);
      float qsc = 1.0f / (nrm * 32.0f);          // /||q|| and /sqrt(D)
      float sal = 0.45f * (tacc[r] * qsc) + cj;
      sal = fminf(fmaxf(sal, 0.0f), 1.0f);
      float m = grp16_max(sal);
      float e = __expf(sal - m);
      float ssum = grp16_sum(e);
      attnL[row * 16 + ln] = e / ssum;
    }
  }
  __syncthreads();

  // ---- phase 4: retrieved = attn @ v -> retr LDS; each wave 1/4 of N, all tiles ----
#if __has_builtin(__builtin_amdgcn_wmma_f32_16x16x4_f32)
  {
    v2f af[MT][4];
#pragma unroll
    for (int tt = 0; tt < MT; ++tt)
#pragma unroll
      for (int kk = 0; kk < 4; ++kk) {
        af[tt][kk][0] = attnL[(tt * 16 + ln) * 16 + kk * 4 + 2 * lh];
        af[tt][kk][1] = attnL[(tt * 16 + ln) * 16 + kk * 4 + 2 * lh + 1];
      }
    for (int n0 = wid * 16; n0 < Dm; n0 += 64) {
      v8f racc[MT];
#pragma unroll
      for (int tt = 0; tt < MT; ++tt) racc[tt] = (v8f){};
#pragma unroll
      for (int kk = 0; kk < 4; ++kk) {
        v2f bf;
        bf[0] = v_ws[(kk * 4 + 2 * lh) * Dm + n0 + ln];
        bf[1] = v_ws[(kk * 4 + 2 * lh + 1) * Dm + n0 + ln];
#pragma unroll
        for (int tt = 0; tt < MT; ++tt)
          racc[tt] = __builtin_amdgcn_wmma_f32_16x16x4_f32(false, af[tt][kk], false, bf, (short)0, racc[tt], false, false);
      }
#pragma unroll
      for (int tt = 0; tt < MT; ++tt)
#pragma unroll
        for (int r = 0; r < 8; ++r)
          retr[(tt * 16 + r + 8 * lh) * Dm + n0 + ln] = (__bf16)racc[tt][r];
    }
  }
#else
  for (int n0 = wid * 16; n0 < Dm; n0 += 64) {
#pragma unroll
    for (int tt = 0; tt < MT; ++tt)
#pragma unroll
      for (int r = 0; r < 8; ++r) {
        float s = 0.f;
#pragma unroll
        for (int j = 0; j < 16; ++j)
          s = fmaf(attnL[(tt * 16 + r + 8 * lh) * 16 + j], v_ws[j * Dm + n0 + ln], s);
        retr[(tt * 16 + r + 8 * lh) * Dm + n0 + ln] = (__bf16)s;
      }
  }
#endif
  __syncthreads();

  // ---- phase 5: gate/out GEMMs (K=2048: x-half + retr-half, both LDS) ----
  {
    v8f hs[MT], hq[MT];
#pragma unroll
    for (int tt = 0; tt < MT; ++tt) { hs[tt] = (v8f){}; hq[tt] = (v8f){}; }
    for (int n0 = wid * 16; n0 < Dm; n0 += 64) {
      v8f ag[MT], ao[MT];
#pragma unroll
      for (int tt = 0; tt < MT; ++tt) { ag[tt] = (v8f){}; ao[tt] = (v8f){}; }
      const __bf16* wg = Wgb + (n0 + ln) * 2048;
      const __bf16* wo = Wob + (n0 + ln) * 2048;
      __builtin_prefetch(wg, 0, 1);
      __builtin_prefetch(wo, 0, 1);
      // K = 0..1023 : A from xt
#pragma unroll 2
      for (int kc = 0; kc < Dm; kc += 32) {
        v16bf b0 = load_bfrag(wg + kc + bkoff);
        v16bf b1 = load_bfrag(wo + kc + bkoff);
#pragma unroll
        for (int tt = 0; tt < MT; ++tt) {
          v16bf a = load_afrag(xt + (tt * 16 + ln) * Dm, kc + akoff);
          ag[tt] = __builtin_amdgcn_wmma_f32_16x16x32_bf16(false, a, false, b0, (short)0, ag[tt], false, false);
          ao[tt] = __builtin_amdgcn_wmma_f32_16x16x32_bf16(false, a, false, b1, (short)0, ao[tt], false, false);
        }
      }
      // K = 1024..2047 : A from retr
#pragma unroll 2
      for (int kc = 0; kc < Dm; kc += 32) {
        v16bf b0 = load_bfrag(wg + Dm + kc + bkoff);
        v16bf b1 = load_bfrag(wo + Dm + kc + bkoff);
#pragma unroll
        for (int tt = 0; tt < MT; ++tt) {
          v16bf a = load_afrag(retr + (tt * 16 + ln) * Dm, kc + akoff);
          ag[tt] = __builtin_amdgcn_wmma_f32_16x16x32_bf16(false, a, false, b0, (short)0, ag[tt], false, false);
          ao[tt] = __builtin_amdgcn_wmma_f32_16x16x32_bf16(false, a, false, b1, (short)0, ao[tt], false, false);
        }
      }
      float bgv = bg[n0 + ln];
      float bov = bo[n0 + ln];
#pragma unroll
      for (int tt = 0; tt < MT; ++tt)
#pragma unroll
        for (int r = 0; r < 8; ++r) {
          int row = tt * 16 + r + 8 * lh;
          int col = n0 + ln;
          float g = 1.0f / (1.0f + __expf(-(ag[tt][r] + bgv)));
          float o = ao[tt][r] + bov;
          float gel = 0.5f * o * (1.0f + erff(o * 0.70710678118654752f));
          float xv = x[(R0 + row) * Dm + col];
          float rv = (float)retr[row * Dm + col];
          float h = gel + g * rv + (1.0f - g) * xv;
          out[(R0 + row) * Dm + col] = h;
          hs[tt][r] += h;
          hq[tt][r] += h * h;
        }
    }
#pragma unroll
    for (int tt = 0; tt < MT; ++tt)
#pragma unroll
      for (int r = 0; r < 8; ++r) {
        float s = grp16_sum(hs[tt][r]);
        float q = grp16_sum(hq[tt][r]);
        if (ln == 0) {
          int row = tt * 16 + r + 8 * lh;
          rhs[wid * 64 + row] = s;
          rhq[wid * 64 + row] = q;
        }
      }
  }
  __syncthreads();

  // ---- phase 6: combine LN stat partials, write per-row mean/rstd ----
  if (tid < 64) {
    float s = rhs[tid] + rhs[64 + tid] + rhs[128 + tid] + rhs[192 + tid];
    float q = rhq[tid] + rhq[64 + tid] + rhq[128 + tid] + rhq[192 + tid];
    float mu = s * (1.0f / Dm);
    float var = q * (1.0f / Dm) - mu * mu;
    long row = R0 + tid;
    stats[2 * row] = mu;
    stats[2 * row + 1] = rsqrtf(var + 1e-5f);
  }
}

// ---------------- LN apply (in place, float4) ----------------
__global__ void wm_ln(float* __restrict__ out, const float* __restrict__ stats,
                      const float* __restrict__ gamma, const float* __restrict__ beta) {
  long idx = (long)blockIdx.x * 256 + threadIdx.x;   // 8M float4s
  long e = idx * 4;
  int row = (int)(e >> 10);
  int col = (int)(e & 1023);
  float mu = stats[2 * row];
  float rs = stats[2 * row + 1];
  float4 h = *(float4*)(out + e);
  float4 gv = *(const float4*)(gamma + col);
  float4 bv = *(const float4*)(beta + col);
  h.x = (h.x - mu) * rs * gv.x + bv.x;
  h.y = (h.y - mu) * rs * gv.y + bv.y;
  h.z = (h.z - mu) * rs * gv.z + bv.z;
  h.w = (h.w - mu) * rs * gv.w + bv.w;
  *(float4*)(out + e) = h;
}

// ---------------- host launcher ----------------
extern "C" void kernel_launch(void* const* d_in, const int* in_sizes, int n_in,
                              void* d_out, int out_size, void* d_ws, size_t ws_size,
                              hipStream_t stream) {
  const float* x        = (const float*)d_in[0];
  const float* buf      = (const float*)d_in[1];
  const float* buf_age  = (const float*)d_in[2];
  const float* buf_freq = (const float*)d_in[3];
  const float* buf_imp  = (const float*)d_in[4];
  const float* Wq       = (const float*)d_in[5];
  const float* Wk       = (const float*)d_in[6];
  const float* Wv       = (const float*)d_in[7];
  const float* Wout     = (const float*)d_in[8];
  const float* bout     = (const float*)d_in[9];
  const float* Wg       = (const float*)d_in[10];
  const float* bg       = (const float*)d_in[11];
  const float* gamma    = (const float*)d_in[12];
  const float* beta     = (const float*)d_in[13];
  float* out = (float*)d_out;

  char* ws = (char*)d_ws;
  size_t off = 0;
  float* v_ws  = (float*)(ws + off);  off += (size_t)CAP * Dm * 4;        // 64KB
  float* k_ws  = (float*)(ws + off);  off += (size_t)CAP * Dm * 4;        // 64KB
  float* salc  = (float*)(ws + off);  off += 256;
  float* stats = (float*)(ws + off);  off += (size_t)Bn * 2 * 4;          // 256KB
  __bf16* Mb   = (__bf16*)(ws + off); off += (size_t)CAP * Dm * 2;        // 32KB
  __bf16* Wqb  = (__bf16*)(ws + off); off += (size_t)Dm * Dm * 2;         // 2MB
  __bf16* Wgb  = (__bf16*)(ws + off); off += (size_t)Dm * 2 * Dm * 2;     // 4MB
  __bf16* Wob  = (__bf16*)(ws + off); off += (size_t)Dm * 2 * Dm * 2;     // 4MB
  (void)ws_size; (void)in_sizes; (void)n_in; (void)out_size;

  wm_prep1<<<128, 256, 0, stream>>>(buf, Wk, Wv, k_ws, v_ws);
  wm_prep2<<<1, 256, 0, stream>>>(buf_age, buf_freq, buf_imp, k_ws, salc);
  wm_prep3<<<64, 256, 0, stream>>>(k_ws, Wq, Mb);
  wm_conv<<<2048, 256, 0, stream>>>(Wq, Wg, Wout, Wqb, Wgb, Wob);
  wm_main<<<NBLOCKS, NWAVE * 32, LDS_BYTES, stream>>>(
      x, v_ws, salc, Mb, Wqb, Wgb, Wob, bg, bout, out, stats);
  wm_ln<<<32768, 256, 0, stream>>>(out, stats, gamma, beta);
}
